// SwinBlock_61366492725558
// MI455X (gfx1250) — matrix operand order repaired and checked by reading, not compiled, not run on hardware
//
#include <hip/hip_runtime.h>

// ============================================================================
// Swin block for MI455X (gfx1250), wave32, bf16 WMMA everywhere.
//
// Pipeline:
//   wT_kernel        : f32 [K][N] weights -> bf16 [N][K] (transposed)
//   ln_kernel<1>     : LN1 + roll(-2,-2) + 5x5 window partition -> wnd bf16
//   gemm_wmma<0>     : qkv = wnd @ qkv_w + b            -> bf16 [160000][768]
//   attn_kernel      : per-window 8-head attention       -> o bf16 [160000][256]
//   gemm_wmma<2>     : x2 = x + unshift(o @ proj_w + b)  -> f32  [160000][256]
//   ln_kernel<0>     : LN2(x2)                           -> h2 bf16
//   gemm_wmma<1>     : a1 = relu(h2 @ fc1_w + b)         -> bf16 [160000][512]
//   gemm_wmma<3>     : out = x2 + a1 @ fc2_w + b         -> f32 (d_out)
//
// GEMM: BM=128, BN=128, BK=32; 8 waves, each owns a 32x64 subtile
// (2x4 v_wmma_f32_16x16x32_bf16 per K-step). B tile staged into LDS by the
// Tensor Data Mover (tensor_load_to_lds + s_wait_tensorcnt) with TDM padding
// matching the 40-element LDS row pitch; A tile staged by vector loads.
//
// Workspace layout (bytes), peak ~575 MB:
//   [0          , 81920000 )  wnd  bf16 160000x256   (reused as h2)
//   [81920000   , 327680000)  qkv  bf16 160000x768   (reused as a1 160000x512)
//   [327680000  , 409600000)  o    bf16 160000x256
//   [409600000  , 573440000)  x2   f32  160000x256
//   [573440000  , ...      )  transposed bf16 weights (~1 MB)
// ============================================================================

typedef __attribute__((ext_vector_type(16))) __bf16        bf16x16;
typedef __attribute__((ext_vector_type(8)))  __bf16        bf16x8;
typedef __attribute__((ext_vector_type(8)))  float         f32x8;
typedef __attribute__((ext_vector_type(4)))  unsigned int  u32x4;
typedef __attribute__((ext_vector_type(8)))  int           i32x8;
typedef __attribute__((ext_vector_type(4)))  int           i32x4;

#define WMMA_BF16(a, b, c) \
  __builtin_amdgcn_wmma_f32_16x16x32_bf16(false, (a), false, (b), (short)0, (c), false, false)

#if defined(__has_builtin)
#if __has_builtin(__builtin_amdgcn_tensor_load_to_lds) && \
    __has_builtin(__builtin_amdgcn_s_wait_tensorcnt)
#define HAVE_TDM 1
#endif
#endif

__device__ __forceinline__ unsigned short f2bf_bits(float f) {
  unsigned u = __float_as_uint(f);
  unsigned r = u + 0x7FFFu + ((u >> 16) & 1u);   // round-to-nearest-even
  return (unsigned short)(r >> 16);
}

__device__ __forceinline__ f32x8 zero_f32x8() {
  f32x8 z;
#pragma unroll
  for (int e = 0; e < 8; ++e) z[e] = 0.f;
  return z;
}

#define SHUF16(lo, hi) \
  __builtin_shufflevector((lo), (hi), 0, 1, 2, 3, 4, 5, 6, 7, 8, 9, 10, 11, 12, 13, 14, 15)

#ifdef HAVE_TDM
// --------------------------------------------------------------------------
// Issue a TDM 2D tile load: tile_d1 rows x tile_d0 elems of 2-byte data from
// global (row pitch stride0 elems) into LDS at lds_byte_addr, inserting 16B
// of LDS padding after every 64B stored (row pitch in LDS = 80B = 40 elems).
// D# layout per CDNA5 ISA 8.3/8.4. Groups 2/3 zero (2D tensor, no iterate);
// this toolchain uses the 6-operand builtin (g0, g1, g2, g3, g2b, cpol).
// --------------------------------------------------------------------------
__device__ __forceinline__ void tdm_load_tile_2d(
    unsigned lds_byte_addr, const void* gaddr,
    unsigned tile_d0, unsigned tile_d1, unsigned long long stride0_elems) {
  unsigned long long ga = (unsigned long long)gaddr;
  u32x4 g0;
  g0[0] = 1u;                                               // count=1, user D#
  g0[1] = lds_byte_addr;                                    // lds_addr
  g0[2] = (unsigned)(ga & 0xFFFFFFFFu);                     // global_addr lo
  g0[3] = (unsigned)((ga >> 32) & 0x01FFFFFFu) | (2u << 30);// addr hi + type=2

  // group1 word0: data_size=1 (2B) | pad_enable | pad_interval=3 (16 dwords)
  //               | pad_amount=3 (4 dwords = 16B)
  unsigned w0 = (1u << 16) | (1u << 20) | (3u << 22) | (3u << 25);
  unsigned long long td0 = tile_d0;   // tensor dims == tile dims (exact tile)
  unsigned long long td1 = tile_d1;
  unsigned long long q0 = (unsigned long long)w0 | ((td0 & 0xFFFFull) << 48);
  unsigned long long q1 = (td0 >> 16) | (td1 << 16) |
                          (((unsigned long long)tile_d0 & 0xFFFFull) << 48);
  unsigned long long q2 = ((unsigned long long)tile_d1 & 0xFFFFull) |
                          ((stride0_elems & 0xFFFFFFFFull) << 32);
  unsigned long long q3 = (stride0_elems >> 32) & 0xFFFFull;
  i32x8 g1;
  g1[0] = (int)(unsigned)q0; g1[1] = (int)(unsigned)(q0 >> 32);
  g1[2] = (int)(unsigned)q1; g1[3] = (int)(unsigned)(q1 >> 32);
  g1[4] = (int)(unsigned)q2; g1[5] = (int)(unsigned)(q2 >> 32);
  g1[6] = (int)(unsigned)q3; g1[7] = (int)(unsigned)(q3 >> 32);
  i32x4 zz4; zz4[0] = 0; zz4[1] = 0; zz4[2] = 0; zz4[3] = 0;
  i32x8 zz8;
#pragma unroll
  for (int e = 0; e < 8; ++e) zz8[e] = 0;
  __builtin_amdgcn_tensor_load_to_lds(g0, g1, zz4, zz4, zz8, 0);
}
#endif

// ---------------------------------------------------------------------------
// Weight transpose + f32->bf16: w [K][N] f32 -> wt [N][K] bf16
// ---------------------------------------------------------------------------
__global__ __launch_bounds__(256) void wT_kernel(const float* __restrict__ w,
                                                 __bf16* __restrict__ wt,
                                                 int K, int N) {
  int idx = blockIdx.x * 256 + threadIdx.x;
  if (idx >= K * N) return;
  int n = idx / K, k = idx % K;
  ((unsigned short*)wt)[idx] = f2bf_bits(w[(size_t)k * N + n]);
}

// ---------------------------------------------------------------------------
// LayerNorm over C=256, one wave per row. PERM=1: output row is in window
// order, source row gathered through roll(-SHIFT) + window partition.
// ---------------------------------------------------------------------------
template <int PERM>
__global__ __launch_bounds__(256) void ln_kernel(const float* __restrict__ x,
                                                 const float* __restrict__ g,
                                                 const float* __restrict__ b,
                                                 __bf16* __restrict__ out) {
  const int lane = threadIdx.x & 31;
  const int wave = threadIdx.x >> 5;
  const int row  = blockIdx.x * 8 + wave;        // [0, 160000)
  int src = row;
  if (PERM) {
    int w = row / 25, t = row % 25;
    int bb = w / 1600, rem = w % 1600;
    int hn = rem / 40, wn = rem % 40;
    int i = t / 5, j = t % 5;
    int hh = hn * 5 + i + 2; if (hh >= 200) hh -= 200;
    int ww = wn * 5 + j + 2; if (ww >= 200) ww -= 200;
    src = bb * 40000 + hh * 200 + ww;
  }
  const float* xr = x + (size_t)src * 256;
  float v[8];
  float s = 0.f;
#pragma unroll
  for (int e = 0; e < 8; ++e) { v[e] = xr[e * 32 + lane]; s += v[e]; }
#pragma unroll
  for (int m = 16; m; m >>= 1) s += __shfl_xor(s, m, 32);
  float mean = s * (1.0f / 256.0f);
  float vs = 0.f;
#pragma unroll
  for (int e = 0; e < 8; ++e) { float d = v[e] - mean; vs += d * d; }
#pragma unroll
  for (int m = 16; m; m >>= 1) vs += __shfl_xor(vs, m, 32);
  float rstd = rsqrtf(vs * (1.0f / 256.0f) + 1e-5f);
  unsigned short* orow = (unsigned short*)out + (size_t)row * 256;
#pragma unroll
  for (int e = 0; e < 8; ++e) {
    int c = e * 32 + lane;
    orow[c] = f2bf_bits((v[e] - mean) * rstd * g[c] + b[c]);
  }
}

// ---------------------------------------------------------------------------
// bf16 WMMA GEMM: C[M][N] = A[M][K] * Bt[N][K]^T + bias, fused epilogues.
// Block = 256 thr (8 waves). Tile BM=128, BN=128, BK=32. Wave -> 32x64 subtile
// (2x4 v_wmma_f32_16x16x32_bf16 per K-step). LDS rows padded to 40 elems
// (80B = 20 banks) to kill bank conflicts on fragment reads. B tile staged by
// the Tensor Data Mover (wave 0 issues, waits TENSORcnt, barrier releases).
// ---------------------------------------------------------------------------
enum { EPI_BIAS_BF16 = 0, EPI_BIAS_RELU_BF16 = 1, EPI_PROJ_RES_F32 = 2, EPI_BIAS_RES_F32 = 3 };

template <int EPI>
__global__ __launch_bounds__(256) void gemm_wmma(const __bf16* __restrict__ A,
                                                 const __bf16* __restrict__ Bt,
                                                 const float* __restrict__ bias,
                                                 __bf16* __restrict__ outBf,
                                                 float* __restrict__ outF,
                                                 const float* __restrict__ resid,
                                                 int M, int N, int K) {
  __shared__ __align__(16) __bf16 As[128][40];
  __shared__ __align__(16) __bf16 Bs[128][40];

  const int tid   = threadIdx.x;
  const int lane  = tid & 31;
  const int wave  = tid >> 5;
  const int wm    = wave & 3;     // 4 M-subtiles of 32
  const int wn    = wave >> 2;    // 2 N-subtiles of 64
  const int m0    = blockIdx.y * 128;
  const int n0    = blockIdx.x * 128;
  const int lhalf = lane >> 4;
  const int l16   = lane & 15;

  f32x8 acc[2][4];
#pragma unroll
  for (int i = 0; i < 2; ++i)
#pragma unroll
    for (int j = 0; j < 4; ++j) acc[i][j] = zero_f32x8();

  for (int k0 = 0; k0 < K; k0 += 32) {
#ifdef HAVE_TDM
    // B tile (128 rows x 32 elems) via Tensor Data Mover, one issue per block.
    if (wave == 0) {
      tdm_load_tile_2d((unsigned)(size_t)&Bs[0][0],
                       Bt + (size_t)n0 * K + k0, 32u, 128u,
                       (unsigned long long)K);
    }
#else
#pragma unroll
    for (int c = 0; c < 2; ++c) {
      int chunk = tid + c * 256;
      int r = chunk >> 2;
      int col = (chunk & 3) * 8;
      *(bf16x8*)&Bs[r][col] = *(const bf16x8*)(Bt + (size_t)(n0 + r) * K + k0 + col);
    }
#endif
    // Stage A tile: 128x32 bf16 = 512 16B chunks, 2 per thread.
#pragma unroll
    for (int c = 0; c < 2; ++c) {
      int chunk = tid + c * 256;
      int r = chunk >> 2;
      int col = (chunk & 3) * 8;
      const __bf16* src = A + (size_t)(m0 + r) * K + k0 + col;
      *(bf16x8*)&As[r][col] = *(const bf16x8*)src;
      if (k0 + 32 < K) __builtin_prefetch(src + 32, 0, 1);   // global_prefetch_b8
    }
#ifdef HAVE_TDM
    if (wave == 0) __builtin_amdgcn_s_wait_tensorcnt(0);
#endif
    __syncthreads();

    // A frag (16x32): lane m = l16, half selects K octets {h*8..} and {16+h*8..}
    bf16x16 af[2];
#pragma unroll
    for (int i = 0; i < 2; ++i) {
      int m = wm * 32 + i * 16 + l16;
      bf16x8 lo = *(const bf16x8*)&As[m][lhalf * 8];
      bf16x8 hi = *(const bf16x8*)&As[m][16 + lhalf * 8];
      af[i] = SHUF16(lo, hi);
    }
    // B frag (32x16): lane n = l16, half selects K octets {h*8..} and {16+h*8..}
    bf16x16 bfr[4];
#pragma unroll
    for (int j = 0; j < 4; ++j) {
      int n = wn * 64 + j * 16 + l16;
      bf16x8 lo = *(const bf16x8*)&Bs[n][lhalf * 8];
      bf16x8 hi = *(const bf16x8*)&Bs[n][16 + lhalf * 8];
      bfr[j] = SHUF16(lo, hi);
    }
#pragma unroll
    for (int i = 0; i < 2; ++i)
#pragma unroll
      for (int j = 0; j < 4; ++j)
        acc[i][j] = WMMA_BF16(af[i], bfr[j], acc[i][j]);
    __syncthreads();
  }

  // Epilogue. C layout: m = vgpr + 8*lhalf, n = l16 within 16x16 tile.
#pragma unroll
  for (int i = 0; i < 2; ++i) {
#pragma unroll
    for (int j = 0; j < 4; ++j) {
      int n = n0 + wn * 64 + j * 16 + l16;
      float bv = bias[n];
#pragma unroll
      for (int v = 0; v < 8; ++v) {
        int m = m0 + wm * 32 + i * 16 + v + 8 * lhalf;
        float val = acc[i][j][v] + bv;
        if (EPI == EPI_BIAS_BF16) {
          ((unsigned short*)outBf)[(size_t)m * N + n] = f2bf_bits(val);
        } else if (EPI == EPI_BIAS_RELU_BF16) {
          ((unsigned short*)outBf)[(size_t)m * N + n] = f2bf_bits(val > 0.f ? val : 0.f);
        } else if (EPI == EPI_PROJ_RES_F32) {
          // window-order row m -> un-partition + roll(+SHIFT) -> spatial row
          int w = m / 25, t = m % 25;
          int bb = w / 1600, rem = w % 1600;
          int hn = rem / 40, wnn = rem % 40;
          int hh = hn * 5 + t / 5 + 2; if (hh >= 200) hh -= 200;
          int ww = wnn * 5 + t % 5 + 2; if (ww >= 200) ww -= 200;
          size_t dst = ((size_t)bb * 40000 + hh * 200 + ww) * 256 + n;
          outF[dst] = resid[dst] + val;
        } else {  // EPI_BIAS_RES_F32
          size_t idx = (size_t)m * N + n;
          outF[idx] = resid[idx] + val;
        }
      }
    }
  }
}

// ---------------------------------------------------------------------------
// Windowed attention: 1 block per window (25 tokens), 1 wave per head (hd=32).
// S = Q K^T (pad 25->32): 4 WMMAs. Softmax rows in LDS (pad cols zeroed so
// padding contributes exactly 0). O = P V: 4 WMMAs. qkv cols: q=[0,256),
// k=[256,512), v=[512,768), head h at offset h*32 (matches reference reshape).
// ---------------------------------------------------------------------------
__global__ __launch_bounds__(256) void attn_kernel(const __bf16* __restrict__ qkv,
                                                   __bf16* __restrict__ o) {
  __shared__ __align__(16) float  Sbuf[8][32][33];
  __shared__ __align__(16) __bf16 Pbuf[8][32][32];
  const int lane  = threadIdx.x & 31;
  const int h     = threadIdx.x >> 5;
  const int wbase = blockIdx.x * 25;
  const int lhalf = lane >> 4;
  const int l16   = lane & 15;
  const float scale = 0.17677669529663687f;   // 32^-0.5

  // Q fragments (A layout), rows clamped for the 25->32 pad.
  bf16x16 aq[2];
#pragma unroll
  for (int ti = 0; ti < 2; ++ti) {
    int m = ti * 16 + l16;
    int row = wbase + (m < 25 ? m : 24);
    const __bf16* p = qkv + (size_t)row * 768 + h * 32;
    aq[ti] = SHUF16(*(const bf16x8*)(p + lhalf * 8),
                    *(const bf16x8*)(p + 16 + lhalf * 8));
  }
  // K^T fragments (B layout: n = token, K-dim = head dim, contiguous).
  bf16x16 bk[2];
#pragma unroll
  for (int tj = 0; tj < 2; ++tj) {
    int n = tj * 16 + l16;
    int row = wbase + (n < 25 ? n : 24);
    const __bf16* p = qkv + (size_t)row * 768 + 256 + h * 32;
    bk[tj] = SHUF16(*(const bf16x8*)(p + lhalf * 8), *(const bf16x8*)(p + 16 + lhalf * 8));
  }
  // S = Q K^T * scale -> LDS
#pragma unroll
  for (int ti = 0; ti < 2; ++ti)
#pragma unroll
    for (int tj = 0; tj < 2; ++tj) {
      f32x8 s = WMMA_BF16(aq[ti], bk[tj], zero_f32x8());
#pragma unroll
      for (int v = 0; v < 8; ++v)
        Sbuf[h][ti * 16 + v + 8 * lhalf][tj * 16 + l16] = s[v] * scale;
    }
  __syncthreads();

  // Row softmax: lane t handles query row t; pad rows/cols -> 0 in P.
  {
    unsigned short* pr = (unsigned short*)&Pbuf[h][lane][0];
    if (lane < 25) {
      float r[25], mx = -3.4e38f;
#pragma unroll
      for (int j = 0; j < 25; ++j) { r[j] = Sbuf[h][lane][j]; mx = fmaxf(mx, r[j]); }
      float sum = 0.f;
#pragma unroll
      for (int j = 0; j < 25; ++j) { r[j] = __expf(r[j] - mx); sum += r[j]; }
      float inv = 1.0f / sum;
#pragma unroll
      for (int j = 0; j < 25; ++j) pr[j] = f2bf_bits(r[j] * inv);
#pragma unroll
      for (int j = 25; j < 32; ++j) pr[j] = 0;
    } else {
#pragma unroll
      for (int j = 0; j < 32; ++j) pr[j] = 0;
    }
  }
  __syncthreads();

  // V fragments (B layout: n = head-dim d, K-dim = token -> strided gather).
  bf16x16 bv[2];
#pragma unroll
  for (int tn = 0; tn < 2; ++tn) {
    int d = tn * 16 + l16;
    const __bf16* pv = qkv + 512 + h * 32 + d;
    bf16x16 f;
#pragma unroll
    for (int e = 0; e < 16; ++e) {
      int k = (e < 8) ? (lhalf * 8 + e) : (16 + lhalf * 8 + (e - 8));
      int row = wbase + (k < 25 ? k : 24);
      f[e] = pv[(size_t)row * 768];
    }
    bv[tn] = f;
  }
  // O = P V, store valid rows as bf16.
#pragma unroll
  for (int ti = 0; ti < 2; ++ti) {
    int m = ti * 16 + l16;
    const __bf16* pp = &Pbuf[h][m][0];
    bf16x16 ap = SHUF16(*(const bf16x8*)(pp + lhalf * 8),
                        *(const bf16x8*)(pp + 16 + lhalf * 8));
#pragma unroll
    for (int tn = 0; tn < 2; ++tn) {
      f32x8 oacc = WMMA_BF16(ap, bv[tn], zero_f32x8());
#pragma unroll
      for (int v = 0; v < 8; ++v) {
        int mm = ti * 16 + v + 8 * lhalf;
        if (mm < 25)
          ((unsigned short*)o)[(size_t)(wbase + mm) * 256 + h * 32 + tn * 16 + l16] =
              f2bf_bits(oacc[v]);
      }
    }
  }
}

// ---------------------------------------------------------------------------
extern "C" void kernel_launch(void* const* d_in, const int* in_sizes, int n_in,
                              void* d_out, int out_size, void* d_ws, size_t ws_size,
                              hipStream_t stream) {
  (void)in_sizes; (void)n_in; (void)out_size; (void)ws_size;
  const float* x      = (const float*)d_in[0];
  const float* n1g    = (const float*)d_in[1];
  const float* n1b    = (const float*)d_in[2];
  const float* qkv_w  = (const float*)d_in[3];
  const float* qkv_b  = (const float*)d_in[4];
  const float* proj_w = (const float*)d_in[5];
  const float* proj_b = (const float*)d_in[6];
  const float* n2g    = (const float*)d_in[7];
  const float* n2b    = (const float*)d_in[8];
  const float* fc1_w  = (const float*)d_in[9];
  const float* fc1_b  = (const float*)d_in[10];
  const float* fc2_w  = (const float*)d_in[11];
  const float* fc2_b  = (const float*)d_in[12];
  float* out = (float*)d_out;

  char* ws = (char*)d_ws;
  __bf16* wnd    = (__bf16*)(ws + 0);
  __bf16* qkv    = (__bf16*)(ws + 81920000);
  __bf16* o_bf   = (__bf16*)(ws + 327680000);
  float*  x2     = (float*) (ws + 409600000);
  __bf16* qkvwT  = (__bf16*)(ws + 573440000);
  __bf16* projwT = qkvwT + 768 * 256;
  __bf16* fc1wT  = projwT + 256 * 256;
  __bf16* fc2wT  = fc1wT + 512 * 256;
  __bf16* h2 = wnd;   // wnd dead after qkv GEMM
  __bf16* a1 = qkv;   // qkv dead after attention

  // Weight transpose + bf16 convert
  wT_kernel<<<(768 * 256 + 255) / 256, 256, 0, stream>>>(qkv_w,  qkvwT, 256, 768);
  wT_kernel<<<(256 * 256 + 255) / 256, 256, 0, stream>>>(proj_w, projwT, 256, 256);
  wT_kernel<<<(512 * 256 + 255) / 256, 256, 0, stream>>>(fc1_w,  fc1wT, 256, 512);
  wT_kernel<<<(256 * 512 + 255) / 256, 256, 0, stream>>>(fc2_w,  fc2wT, 512, 256);

  // LN1 fused with shift + window partition
  ln_kernel<1><<<20000, 256, 0, stream>>>(x, n1g, n1b, wnd);

  // QKV projection
  gemm_wmma<EPI_BIAS_BF16><<<dim3(6, 1250), 256, 0, stream>>>(
      wnd, qkvwT, qkv_b, qkv, nullptr, nullptr, 160000, 768, 256);

  // Windowed attention
  attn_kernel<<<6400, 256, 0, stream>>>(qkv, o_bf);

  // Proj + un-shift scatter + residual -> x2 (f32)
  gemm_wmma<EPI_PROJ_RES_F32><<<dim3(2, 1250), 256, 0, stream>>>(
      o_bf, projwT, proj_b, nullptr, x2, x, 160000, 256, 256);

  // LN2
  ln_kernel<0><<<20000, 256, 0, stream>>>(x2, n2g, n2b, h2);

  // MLP fc1 + ReLU
  gemm_wmma<EPI_BIAS_RELU_BF16><<<dim3(4, 1250), 256, 0, stream>>>(
      h2, fc1wT, fc1_b, a1, nullptr, nullptr, 160000, 512, 256);

  // MLP fc2 + residual -> d_out (f32)
  gemm_wmma<EPI_BIAS_RES_F32><<<dim3(2, 1250), 256, 0, stream>>>(
      a1, fc2wT, fc2_b, nullptr, out, x2, 160000, 256, 512);
}
